// EncoderTact_88364657148601
// MI455X (gfx1250) — matrix operand order repaired and checked by reading, not compiled, not run on hardware
//
#include <hip/hip_runtime.h>

// ---------------------------------------------------------------------------
// SLAYER SRMALPHA layer for MI455X (gfx1250, wave32, WMMA).
// GEMM runs on raw binary spikes (exact in bf16) because the alpha filter and
// the dense layer commute; the two alpha recurrences (PSP + refractory spike)
// are fused into one sequential scan. GEMM uses v_wmma_f32_16x16x32_bf16 with
// double-buffered LDS fed by GLOBAL_LOAD_ASYNC_TO_LDS_B128 (ASYNCcnt path).
// ---------------------------------------------------------------------------

typedef __bf16 bf16_t;
typedef bf16_t v16bf __attribute__((ext_vector_type(16)));
typedef float  v8f   __attribute__((ext_vector_type(8)));

#define BM 128
#define BN 128
#define BK 32
#define LDW 48   // LDS row pitch in bf16 (BK + 16 pad; 16B-aligned rows, fewer bank conflicts)

// ---- CDNA5 async global->LDS plumbing (guarded; falls back to sync copy) ---
// Builtin signature (from hipcc diagnostic): param0 = int __vector(4) * in the
// global (__device__/AS1) space, param1 = LDS (AS3), then imm offset, imm cpol.
typedef int v4i_ __attribute__((vector_size(16)));
typedef __attribute__((address_space(1))) v4i_ g_v4i;
typedef __attribute__((address_space(3))) v4i_ l_v4i;

#if defined(__has_builtin)
#  if __has_builtin(__builtin_amdgcn_global_load_async_to_lds_b128)
#    define HAVE_ASYNC_LDS 1
#  endif
#endif

#ifdef HAVE_ASYNC_LDS
#  define ASYNC_CP16(g, l)                                                     \
     __builtin_amdgcn_global_load_async_to_lds_b128(                           \
         (g_v4i*)(void*)(g), (l_v4i*)(void*)(l), 0, 0)
#  if __has_builtin(__builtin_amdgcn_s_wait_asynccnt)
#    define WAIT_ASYNC(n) __builtin_amdgcn_s_wait_asynccnt(n)
#  else
#    define WAIT_ASYNC(n) asm volatile("s_wait_asynccnt " #n ::: "memory")
#  endif
#else
#  define ASYNC_CP16(g, l) (*(uint4*)(l) = *(const uint4*)(g))
#  define WAIT_ASYNC(n) ((void)0)
#endif

// ------------------------- W: f32 -> bf16 ----------------------------------
__global__ void cast_w(const float* __restrict__ W, bf16_t* __restrict__ Wb, int n) {
    int i = blockIdx.x * blockDim.x + threadIdx.x;
    if (i < n) Wb[i] = (bf16_t)W[i];
}

// --------------- spikes [B,C,T] f32 -> A [T*B, C] bf16 ---------------------
__global__ void transpose_cast(const float* __restrict__ x, bf16_t* __restrict__ A,
                               int Bsz, int C, int T, int K) {
    __shared__ float tile[32][33];
    int t0 = blockIdx.x * 32, c0 = blockIdx.y * 32, b = blockIdx.z;
    int tx = threadIdx.x, ty = threadIdx.y;
    #pragma unroll
    for (int i = 0; i < 32; i += 8) {
        int c = c0 + ty + i, t = t0 + tx;
        tile[tx][ty + i] = (t < T) ? x[((size_t)b * C + c) * T + t] : 0.0f;
    }
    __syncthreads();
    #pragma unroll
    for (int i = 0; i < 32; i += 8) {
        int t = t0 + ty + i, c = c0 + tx;
        if (t < T) A[((size_t)t * Bsz + b) * K + c] = (bf16_t)tile[ty + i][tx];
    }
}

// ------------------------- WMMA bf16 GEMM ----------------------------------
// A-fragment (16x32 bf16), ISA 7.12.2: lanes 0-15 row M=lane, K=0..7 in v0-3,
// K=16..23 in v4-7; lanes 16-31 same rows, K=8..15 / K=24..31.
__device__ __forceinline__ v16bf ldfragA(const bf16_t* base, int lane) {
    int row  = lane & 15;
    int ksel = (lane >> 4) & 1;
    const bf16_t* p = base + row * LDW + ksel * 8;
    union { uint4 u[2]; v16bf v; } r;
    r.u[0] = *(const uint4*)(p);        // K 0..7   (or 8..15)
    r.u[1] = *(const uint4*)(p + 16);   // K 16..23 (or 24..31)
    return r.v;
}

// B-fragment (32x16 bf16, supplied as N-major rows of W): lanes 0-15 column
// N=lane hold K=0..7 in v0-3, K=16..23 in v4-7; lanes 16-31 K=8..15 / K=24..31.
__device__ __forceinline__ v16bf ldfragB(const bf16_t* base, int lane) {
    int col  = lane & 15;
    int ksel = (lane >> 4) & 1;
    const bf16_t* p = base + col * LDW + ksel * 8;
    union { uint4 u[2]; v16bf v; } r;
    r.u[0] = *(const uint4*)(p);
    r.u[1] = *(const uint4*)(p + 16);
    return r.v;
}

__global__ __launch_bounds__(256)
void gemm_bf16(const bf16_t* __restrict__ A, const bf16_t* __restrict__ Bw,
               float* __restrict__ U, int M, int N, int K) {
    __shared__ bf16_t As[2][BM * LDW];   // 2 x 12 KB
    __shared__ bf16_t Bs[2][BN * LDW];   // 2 x 12 KB

    int tid  = threadIdx.x;
    int lane = tid & 31;
    int wave = tid >> 5;          // 8 waves
    int wr   = wave & 1;          // 2 wave-rows  (64 m each)
    int wc   = wave >> 1;         // 4 wave-cols  (32 n each)
    int m0   = blockIdx.x * BM;
    int n0   = blockIdx.y * BN;

    // per-thread 16B staging chunk: row = tid>>2 (+64), seg = tid&3
    int row = tid >> 2, seg = tid & 3;
    const bf16_t* gA0 = A  + (size_t)(m0 + row) * K + seg * 8;
    const bf16_t* gA1 = gA0 + (size_t)64 * K;
    const bf16_t* gB0 = Bw + (size_t)(n0 + row) * K + seg * 8;
    const bf16_t* gB1 = gB0 + (size_t)64 * K;
    int lOff0 = row * LDW + seg * 8;
    int lOff1 = lOff0 + 64 * LDW;

#define ISSUE(buf, kc)                              \
    do {                                            \
        ASYNC_CP16(gA0 + (kc), &As[buf][lOff0]);    \
        ASYNC_CP16(gA1 + (kc), &As[buf][lOff1]);    \
        ASYNC_CP16(gB0 + (kc), &Bs[buf][lOff0]);    \
        ASYNC_CP16(gB1 + (kc), &Bs[buf][lOff1]);    \
    } while (0)

    v8f acc[4][2];
    #pragma unroll
    for (int i = 0; i < 4; i++)
        #pragma unroll
        for (int j = 0; j < 2; j++) acc[i][j] = (v8f){};

    int nit = K / BK;
    ISSUE(0, 0);                               // prologue: group 0 -> buf 0

    for (int i = 0; i < nit; i++) {
        int cur = i & 1;
        if (i + 1 < nit) {                     // issue-ahead group i+1
            ISSUE(1 - cur, (i + 1) * BK);
            WAIT_ASYNC(4);                     // retire group i (older 4 loads)
        } else {
            WAIT_ASYNC(0);
        }
        __syncthreads();                       // all waves' group i visible

        v16bf af[4], bfg[2];
        #pragma unroll
        for (int mt = 0; mt < 4; mt++)
            af[mt] = ldfragA(&As[cur][(wr * 64 + mt * 16) * LDW], lane);
        #pragma unroll
        for (int nt = 0; nt < 2; nt++)
            bfg[nt] = ldfragB(&Bs[cur][(wc * 32 + nt * 16) * LDW], lane);

        #pragma unroll
        for (int mt = 0; mt < 4; mt++)
            #pragma unroll
            for (int nt = 0; nt < 2; nt++)
                acc[mt][nt] = __builtin_amdgcn_wmma_f32_16x16x32_bf16(
                    false, af[mt], false, bfg[nt], (short)0, acc[mt][nt],
                    false, false);
        __syncthreads();                       // buffer safe for reuse
    }
#undef ISSUE

    // C/D layout: VGPR r -> (M = r + 8*(lane>=16), N = lane & 15)
    int mgrp = (lane >> 4) * 8;
    int ncol = lane & 15;
    #pragma unroll
    for (int mt = 0; mt < 4; mt++)
        #pragma unroll
        for (int nt = 0; nt < 2; nt++) {
            size_t mb = (size_t)m0 + wr * 64 + mt * 16 + mgrp;
            size_t nb = (size_t)n0 + wc * 32 + nt * 16 + ncol;
            #pragma unroll
            for (int r = 0; r < 8; r++)
                U[(mb + r) * (size_t)N + nb] = acc[mt][nt][r];
        }
}

// --------------- fused PSP alpha filter + refractory spike scan ------------
__global__ __launch_bounds__(256)
void snn_scan(const float* __restrict__ U, float* __restrict__ out,
              int Bsz, int O, int T) {
    int g = blockIdx.x * blockDim.x + threadIdx.x;  // 0 .. B*O-1
    int b = g / O, o = g - b * O;

    const float a1 = 0.9048374180359596f;    // exp(-TS/tau_sr)
    const float c1 = 0.2718281828459045f;    // e*TS/tau_sr
    const float a2 = 0.36787944117144233f;   // exp(-TS/tau_ref)
    const float c2 = -54.365636569180902f;   // -scaleRef*theta*e*TS/tau_ref

    float p1 = 0.f, q1 = 0.f, p2 = 0.f, q2 = 0.f;
    float4 sbuf;
    float* srow = out + ((size_t)b * O + o) * T;

    for (int t = 0; t < T; t++) {
        float v = U[((size_t)t * Bsz + b) * O + o];
        q1 = a1 * (q1 + p1);
        p1 = a1 * p1 + v;
        float u = c1 * q1;                    // membrane potential
        q2 = a2 * (q2 + p2);                  // refractory trace
        float s = (u + c2 * q2 >= 10.0f) ? 1.0f : 0.0f;
        p2 = a2 * p2 + s;
        ((float*)&sbuf)[t & 3] = s;
        if ((t & 3) == 3)
            *(float4*)(srow + (t - 3)) = sbuf;   // T=500 divisible by 4
    }
}

// ---------------------------------------------------------------------------
extern "C" void kernel_launch(void* const* d_in, const int* in_sizes, int n_in,
                              void* d_out, int out_size, void* d_ws, size_t ws_size,
                              hipStream_t stream) {
    const float* x = (const float*)d_in[0];   // [B, C, 1, 1, T] f32 spikes
    const float* W = (const float*)d_in[1];   // [O, C] f32
    float* out = (float*)d_out;               // [B, O, 1, 1, T] f32

    const int Bsz = 32, C = 1024, O = 1024, T = 500;
    const int M = T * Bsz, K = C, N = O;

    char* ws = (char*)d_ws;
    bf16_t* Abf = (bf16_t*)ws;                       // 32 MB
    size_t aBytes = (size_t)M * K * sizeof(bf16_t);
    bf16_t* Wbf = (bf16_t*)(ws + aBytes);            // 2 MB
    size_t wBytes = (size_t)N * K * sizeof(bf16_t);
    float* U = (float*)(ws + aBytes + wBytes);       // 64 MB

    cast_w<<<(N * K + 255) / 256, 256, 0, stream>>>(W, Wbf, N * K);

    dim3 tg((T + 31) / 32, C / 32, Bsz);
    transpose_cast<<<tg, dim3(32, 8), 0, stream>>>(x, Abf, Bsz, C, T, K);

    dim3 gg(M / BM, N / BN);                         // 125 x 8
    gemm_bf16<<<gg, 256, 0, stream>>>(Abf, Wbf, U, M, N, K);

    snn_scan<<<(Bsz * O) / 256, 256, 0, stream>>>(U, out, Bsz, O, T);
}
